// LSM_50096498540854
// MI455X (gfx1250) — compile-verified, matrix operand order checked by reading, not run on hardware
//
#include <hip/hip_runtime.h>
#include <hip/hip_bf16.h>
#include <math.h>
#include <stdint.h>

typedef __attribute__((ext_vector_type(16))) __bf16 v16bf;
typedef __attribute__((ext_vector_type(8)))  float  v8f;

#ifndef __has_builtin
#define __has_builtin(x) 0
#endif

#if __has_builtin(__builtin_amdgcn_tensor_load_to_lds) && __has_builtin(__builtin_amdgcn_s_wait_tensorcnt)
#define USE_TDM 1
#else
#define USE_TDM 0
#endif

#define LAYERS 3
#define N1     4096
#define N2     4096
#define DD     1024
#define KK     10
#define FDIM   2048      // 1024 full level + 1022 pooled levels + 2 pad
#define EPSV   1e-6f

#define BM  128
#define BN  128
#define BKC 64
#define LDT 72           // padded LDS stride (elements) = 64 + 8 -> conflict-free b128 reads

__device__ __forceinline__ float softplusf(float x) {
    return x > 20.f ? x : log1pf(expf(x));
}

#if USE_TDM
typedef __attribute__((ext_vector_type(4))) unsigned tdm_v4u;
typedef __attribute__((ext_vector_type(8))) int      tdm_v8i;
typedef __attribute__((ext_vector_type(4))) int      tdm_v4i;

// TDM descriptor: 2D tensor (FDIM x 4096) bf16, tile 64 x 128, LDS padding
// 4 DWORDs after every 32 DWORDs => LDS row stride 144B == LDT(72) elements.
__device__ __forceinline__ void tdm_load_tile(unsigned lds_byte_addr,
                                              unsigned long long gaddr)
{
    tdm_v4u g0;
    g0[0] = 1u;                                        // count=1, user descriptor
    g0[1] = lds_byte_addr;                             // lds_addr [63:32]
    g0[2] = (unsigned)(gaddr & 0xffffffffu);           // global_addr lo
    g0[3] = (unsigned)((gaddr >> 32) & 0x01ffffffu)    // global_addr hi (bits 120:96 region)
          | (2u << 30);                                // type = 2 (image)
    tdm_v8i g1;
    g1[0] = (int)((1u << 16)      // data_size = 1 (2 bytes)
          |       (1u << 20)      // pad_enable
          |       (4u << 22)      // pad_interval: 32 DWORDs (128B row)
          |       (3u << 25));    // pad_amount:   4 DWORDs (16B)
    g1[1] = (int)(((unsigned)FDIM & 0xffffu) << 16);   // tensor_dim0 lo16 (bits 63:48)
    g1[2] = (int)(((unsigned)FDIM >> 16) & 0xffffu)    // tensor_dim0 hi16
          | (int)((4096u & 0xffffu) << 16);            // tensor_dim1 lo16
    g1[3] = (int)((4096u >> 16) & 0xffffu)             // tensor_dim1 hi16
          | (int)(64u << 16);                          // tile_dim0 = 64
    g1[4] = (int)128u;                                 // tile_dim1 = 128, tile_dim2 = 0
    g1[5] = (int)FDIM;                                 // tensor_dim0_stride lo32
    g1[6] = 0;                                         // stride0 hi16 | stride1 lo16 (=0)
    g1[7] = (int)(((unsigned long long)FDIM * 4096ull) >> 16); // tensor_dim1_stride[47:16]
    tdm_v4i g2; g2[0] = 1; g2[1] = 0; g2[2] = 0; g2[3] = 0;    // tensor_dim2 = 1
    tdm_v4i g3; g3[0] = 0; g3[1] = 0; g3[2] = 0; g3[3] = 0;
#if __clang_major__ >= 23
    tdm_v8i g4; g4[0]=0; g4[1]=0; g4[2]=0; g4[3]=0; g4[4]=0; g4[5]=0; g4[6]=0; g4[7]=0;
    __builtin_amdgcn_tensor_load_to_lds(g0, g1, g2, g3, g4, 0);
#else
    __builtin_amdgcn_tensor_load_to_lds(g0, g1, g2, g3, 0);
#endif
}
#endif // USE_TDM

// ---------------------------------------------------------------------------
// Feature builder for Z side: softmax row -> [z | pool1 | ... | pool9 | 0,0]
// also a[n] = softmax(latent_z1[n])[layer] * softplus(L), g[n] = gamma[n,layer]
// ---------------------------------------------------------------------------
__global__ __launch_bounds__(256) void feat_z_kernel(
    const float* __restrict__ us_layer, const float* __restrict__ latent_z1,
    const float* __restrict__ gamma, const float* __restrict__ L_,
    __hip_bfloat16* __restrict__ Zf, float* __restrict__ avec,
    float* __restrict__ gvec, int layer)
{
    __shared__ float sbuf[DD];
    __shared__ float pbuf[DD / 2];
    __shared__ float red[256];
    const int n   = blockIdx.x;
    const int tid = threadIdx.x;
    const float* row = us_layer + (size_t)n * DD;

    float x[4];
    float lmax = -3.4e38f;
    #pragma unroll
    for (int t = 0; t < 4; ++t) { x[t] = row[tid + t * 256]; lmax = fmaxf(lmax, x[t]); }
    red[tid] = lmax; __syncthreads();
    for (int s = 128; s > 0; s >>= 1) { if (tid < s) red[tid] = fmaxf(red[tid], red[tid + s]); __syncthreads(); }
    const float m = red[0]; __syncthreads();

    float lsum = 0.f;
    #pragma unroll
    for (int t = 0; t < 4; ++t) { x[t] = expf(x[t] - m); lsum += x[t]; }
    red[tid] = lsum; __syncthreads();
    for (int s = 128; s > 0; s >>= 1) { if (tid < s) red[tid] += red[tid + s]; __syncthreads(); }
    const float inv = 1.f / red[0]; __syncthreads();

    __hip_bfloat16* zr = Zf + (size_t)n * FDIM;
    #pragma unroll
    for (int t = 0; t < 4; ++t) {
        float z = x[t] * inv;
        sbuf[tid + t * 256] = z;
        zr[tid + t * 256] = __float2bfloat16(z);
    }
    __syncthreads();

    // hierarchical 2x pooling: levels 512,256,...,2
    float* src = sbuf;
    float* dst = pbuf;
    int len = DD, off = DD;
    for (int lev = 1; lev < KK; ++lev) {
        const int half = len >> 1;
        for (int j = tid; j < half; j += 256) {
            float v = src[2 * j] + src[2 * j + 1];
            dst[j] = v;
            zr[off + j] = __float2bfloat16(v);
        }
        __syncthreads();
        off += half; len = half;
        float* tmp = src; src = dst; dst = tmp;
    }
    if (tid < 2) zr[2046 + tid] = __float2bfloat16(0.f);

    if (tid == 0) {
        float l0 = latent_z1[n * LAYERS + 0];
        float l1 = latent_z1[n * LAYERS + 1];
        float l2 = latent_z1[n * LAYERS + 2];
        float mm = fmaxf(l0, fmaxf(l1, l2));
        float e0 = expf(l0 - mm), e1 = expf(l1 - mm), e2 = expf(l2 - mm);
        float s  = e0 + e1 + e2;
        float sel = (layer == 0 ? e0 : (layer == 1 ? e1 : e2)) / s;
        avec[n] = sel * softplusf(L_[0]);
        gvec[n] = gamma[n * LAYERS + layer];
    }
}

// ---------------------------------------------------------------------------
// Feature builder for W side: p_k and +EPS folded in:
//  [p_K*(w+eps) | p_1*(pool1+eps) | ... | p_9*(pool9+eps) | 0,0]
// ---------------------------------------------------------------------------
__global__ __launch_bounds__(256) void feat_w_kernel(
    const float* __restrict__ vs_layer, const float* __restrict__ latent_w1,
    const float* __restrict__ delta, const float* __restrict__ p_ks,
    __hip_bfloat16* __restrict__ Wf, float* __restrict__ bvec,
    float* __restrict__ dvec, int layer)
{
    __shared__ float sbuf[DD];
    __shared__ float pbuf[DD / 2];
    __shared__ float red[256];
    __shared__ float pk[KK + 1];
    const int n   = blockIdx.x;
    const int tid = threadIdx.x;
    const float* row = vs_layer + (size_t)n * DD;

    float x[4];
    float lmax = -3.4e38f;
    #pragma unroll
    for (int t = 0; t < 4; ++t) { x[t] = row[tid + t * 256]; lmax = fmaxf(lmax, x[t]); }
    if (tid == 0) {  // softmax over p_ks[layer, 0..K]
        float mx = -3.4e38f;
        for (int i = 0; i <= KK; ++i) mx = fmaxf(mx, p_ks[layer * (KK + 1) + i]);
        float s = 0.f;
        for (int i = 0; i <= KK; ++i) { pk[i] = expf(p_ks[layer * (KK + 1) + i] - mx); s += pk[i]; }
        for (int i = 0; i <= KK; ++i) pk[i] /= s;
    }
    red[tid] = lmax; __syncthreads();
    for (int s = 128; s > 0; s >>= 1) { if (tid < s) red[tid] = fmaxf(red[tid], red[tid + s]); __syncthreads(); }
    const float m = red[0]; __syncthreads();

    float lsum = 0.f;
    #pragma unroll
    for (int t = 0; t < 4; ++t) { x[t] = expf(x[t] - m); lsum += x[t]; }
    red[tid] = lsum; __syncthreads();
    for (int s = 128; s > 0; s >>= 1) { if (tid < s) red[tid] += red[tid + s]; __syncthreads(); }
    const float inv = 1.f / red[0]; __syncthreads();

    __hip_bfloat16* wr = Wf + (size_t)n * FDIM;
    const float pKtop = pk[KK];
    #pragma unroll
    for (int t = 0; t < 4; ++t) {
        float z = x[t] * inv;
        sbuf[tid + t * 256] = z;                              // unscaled for pooling
        wr[tid + t * 256] = __float2bfloat16(pKtop * (z + EPSV));
    }
    __syncthreads();

    float* src = sbuf;
    float* dst = pbuf;
    int len = DD, off = DD;
    for (int lev = 1; lev < KK; ++lev) {
        const int half = len >> 1;
        const float pl = pk[lev];
        for (int j = tid; j < half; j += 256) {
            float v = src[2 * j] + src[2 * j + 1];
            dst[j] = v;
            wr[off + j] = __float2bfloat16(pl * (v + EPSV));
        }
        __syncthreads();
        off += half; len = half;
        float* tmp = src; src = dst; dst = tmp;
    }
    if (tid < 2) wr[2046 + tid] = __float2bfloat16(0.f);

    if (tid == 0) {
        float l0 = latent_w1[n * LAYERS + 0];
        float l1 = latent_w1[n * LAYERS + 1];
        float l2 = latent_w1[n * LAYERS + 2];
        float mm = fmaxf(l0, fmaxf(l1, l2));
        float e0 = expf(l0 - mm), e1 = expf(l1 - mm), e2 = expf(l2 - mm);
        float s  = e0 + e1 + e2;
        float sel = (layer == 0 ? e0 : (layer == 1 ? e1 : e2)) / s;
        bvec[n] = sel + EPSV;
        dvec[n] = delta[n * LAYERS + layer];
    }
}

// ---------------------------------------------------------------------------
// Fused WMMA GEMM (128x128x2048 bf16, f32 acc) + softplus epilogue.
// TDM path: double-buffered LDS; wave0 drives tensor_load_to_lds DMAs for
// chunk k+1 (TENSORcnt) while all 8 waves run WMMAs on chunk k.
// ---------------------------------------------------------------------------
__global__ __launch_bounds__(256) void gemm_softplus_kernel(
    const __hip_bfloat16* __restrict__ Zf, const __hip_bfloat16* __restrict__ Wf,
    const float* __restrict__ avec, const float* __restrict__ bvec,
    const float* __restrict__ gvec, const float* __restrict__ dvec,
    const float* __restrict__ p_ks, int layer, float* __restrict__ part)
{
    __shared__ float red[256];

    const int tid  = threadIdx.x;
    const int lane = tid & 31;
    const int wid  = tid >> 5;
    const int wm   = wid & 3;      // 4 wave-rows  x 32 rows
    const int wn   = wid >> 2;     // 2 wave-cols  x 64 cols
    const int m0   = blockIdx.y * BM;
    const int n0   = blockIdx.x * BN;

    // p0 = softmax(p_ks[layer])[0]  (uniform -> scalarized)
    float p0;
    {
        float mx = -3.4e38f;
        for (int i = 0; i <= KK; ++i) mx = fmaxf(mx, p_ks[layer * (KK + 1) + i]);
        float s = 0.f;
        for (int i = 0; i <= KK; ++i) s += expf(p_ks[layer * (KK + 1) + i] - mx);
        p0 = expf(p_ks[layer * (KK + 1)] - mx) / s;
    }

    v8f acc[2][4];
    #pragma unroll
    for (int t = 0; t < 2; ++t)
        #pragma unroll
        for (int u = 0; u < 4; ++u)
            #pragma unroll
            for (int i = 0; i < 8; ++i) acc[t][u][i] = 0.f;

    const int khalf = (lane >> 4); // K-half select per WMMA 16-bit layout
    const int l15   = lane & 15;

#if USE_TDM
    __shared__ __align__(16) __hip_bfloat16 As[2][BM * LDT];
    __shared__ __align__(16) __hip_bfloat16 Bs[2][BN * LDT];

    const unsigned long long zg = (unsigned long long)(uintptr_t)Zf;
    const unsigned long long wg = (unsigned long long)(uintptr_t)Wf;
    // LDS byte addresses: generic-pointer low 32 bits == LDS offset (ISA 10.2)
    const unsigned asAddr[2] = { (unsigned)(uintptr_t)&As[0][0], (unsigned)(uintptr_t)&As[1][0] };
    const unsigned bsAddr[2] = { (unsigned)(uintptr_t)&Bs[0][0], (unsigned)(uintptr_t)&Bs[1][0] };

    if (wid == 0) {
        tdm_load_tile(asAddr[0], zg + ((unsigned long long)m0 * FDIM) * 2ull);
        tdm_load_tile(bsAddr[0], wg + ((unsigned long long)n0 * FDIM) * 2ull);
    }

    for (int kc = 0; kc < FDIM; kc += BKC) {
        const int buf = (kc / BKC) & 1;
        if (wid == 0) __builtin_amdgcn_s_wait_tensorcnt(0);   // chunk kc resident
        __syncthreads();                                       // release all waves
        if (wid == 0 && (kc + BKC) < FDIM) {                   // prefetch chunk kc+1
            tdm_load_tile(asAddr[buf ^ 1], zg + ((unsigned long long)m0 * FDIM + kc + BKC) * 2ull);
            tdm_load_tile(bsAddr[buf ^ 1], wg + ((unsigned long long)n0 * FDIM + kc + BKC) * 2ull);
        }
        const __hip_bfloat16* Ab = &As[buf][0];
        const __hip_bfloat16* Bb = &Bs[buf][0];
        #pragma unroll
        for (int kk = 0; kk < BKC; kk += 32) {
            const int kb = kk + khalf * 8;
            union { uint4 q[2]; v16bf v; } fa[2], fb[4];
            #pragma unroll
            for (int t = 0; t < 2; ++t) {
                const int r = wm * 32 + t * 16 + l15;
                fa[t].q[0] = *(const uint4*)(Ab + r * LDT + kb);
                fa[t].q[1] = *(const uint4*)(Ab + r * LDT + kb + 16);
            }
            #pragma unroll
            for (int u = 0; u < 4; ++u) {
                const int c = wn * 64 + u * 16 + l15;
                fb[u].q[0] = *(const uint4*)(Bb + c * LDT + kb);
                fb[u].q[1] = *(const uint4*)(Bb + c * LDT + kb + 16);
            }
            #pragma unroll
            for (int t = 0; t < 2; ++t)
                #pragma unroll
                for (int u = 0; u < 4; ++u)
                    acc[t][u] = __builtin_amdgcn_wmma_f32_16x16x32_bf16(
                        false, fa[t].v, false, fb[u].v, (short)0, acc[t][u], false, false);
        }
    }
#else
    __shared__ __align__(16) __hip_bfloat16 As[BM * LDT];
    __shared__ __align__(16) __hip_bfloat16 Bs[BN * LDT];
    const int lrow = tid >> 3;     // 0..31
    const int lseg = tid & 7;      // 0..7  (16B segments)

    for (int kc = 0; kc < FDIM; kc += BKC) {
        __syncthreads();
        #pragma unroll
        for (int r = 0; r < 4; ++r) {
            const int row = lrow + r * 32;
            uint4 va = *(const uint4*)(Zf + (size_t)(m0 + row) * FDIM + kc + lseg * 8);
            *(uint4*)(&As[row * LDT + lseg * 8]) = va;
            uint4 vb = *(const uint4*)(Wf + (size_t)(n0 + row) * FDIM + kc + lseg * 8);
            *(uint4*)(&Bs[row * LDT + lseg * 8]) = vb;
        }
        __syncthreads();
        #pragma unroll
        for (int kk = 0; kk < BKC; kk += 32) {
            const int kb = kk + khalf * 8;
            union { uint4 q[2]; v16bf v; } fa[2], fb[4];
            #pragma unroll
            for (int t = 0; t < 2; ++t) {
                const int r = wm * 32 + t * 16 + l15;
                fa[t].q[0] = *(const uint4*)(&As[r * LDT + kb]);
                fa[t].q[1] = *(const uint4*)(&As[r * LDT + kb + 16]);
            }
            #pragma unroll
            for (int u = 0; u < 4; ++u) {
                const int c = wn * 64 + u * 16 + l15;
                fb[u].q[0] = *(const uint4*)(&Bs[c * LDT + kb]);
                fb[u].q[1] = *(const uint4*)(&Bs[c * LDT + kb + 16]);
            }
            #pragma unroll
            for (int t = 0; t < 2; ++t)
                #pragma unroll
                for (int u = 0; u < 4; ++u)
                    acc[t][u] = __builtin_amdgcn_wmma_f32_16x16x32_bf16(
                        false, fa[t].v, false, fb[u].v, (short)0, acc[t][u], false, false);
        }
    }
#endif

    // epilogue: C/D layout -> M = i + 8*(lane>>4), N = lane&15
    float local = 0.f;
    const int mwBase = m0 + wm * 32;
    const int nwBase = n0 + wn * 64;
    const int mhi = khalf * 8;
    #pragma unroll
    for (int t = 0; t < 2; ++t) {
        #pragma unroll
        for (int i = 0; i < 8; ++i) {
            const int gr = mwBase + t * 16 + mhi + i;
            const float av = avec[gr];
            const float gv = gvec[gr];
            #pragma unroll
            for (int u = 0; u < 4; ++u) {
                const int gc = nwBase + u * 16 + l15;
                if (gr == gc) continue;   // implements sum - trace
                const float inter = p0 + acc[t][u][i];
                const float xv = av * bvec[gc] * inter + gv + dvec[gc];
                local += softplusf(xv);
            }
        }
    }

    red[tid] = local; __syncthreads();
    for (int s = 128; s > 0; s >>= 1) { if (tid < s) red[tid] += red[tid + s]; __syncthreads(); }
    if (tid == 0) part[blockIdx.y * gridDim.x + blockIdx.x] = -red[0];  // -z_pdist1 part
}

// ---------------------------------------------------------------------------
// Sparse term: one wave per (i,j) pair; 2048-length bf16 dot (L2-resident).
// ---------------------------------------------------------------------------
__global__ __launch_bounds__(256) void sparse_kernel(
    const __hip_bfloat16* __restrict__ Zf, const __hip_bfloat16* __restrict__ Wf,
    const float* __restrict__ avec, const float* __restrict__ bvec,
    const float* __restrict__ gvec, const float* __restrict__ dvec,
    const int* __restrict__ is_, const int* __restrict__ js_,
    const float* __restrict__ p_ks, int layer, int E, float* __restrict__ part)
{
    __shared__ float wred[8];
    __shared__ float s_p0;
    const int tid  = threadIdx.x;
    const int lane = tid & 31;
    const int wid  = tid >> 5;

    if (tid == 0) {
        float mx = -3.4e38f;
        for (int i = 0; i <= KK; ++i) mx = fmaxf(mx, p_ks[layer * (KK + 1) + i]);
        float s = 0.f;
        for (int i = 0; i <= KK; ++i) s += expf(p_ks[layer * (KK + 1) + i] - mx);
        s_p0 = expf(p_ks[layer * (KK + 1)] - mx) / s;
    }
    if (tid < 8) wred[tid] = 0.f;
    __syncthreads();

    const int e = blockIdx.x * 8 + wid;
    if (e < E) {
        const int i = is_[e];
        const int j = js_[e];
        const unsigned* zr = (const unsigned*)(Zf + (size_t)i * FDIM);
        const unsigned* wr = (const unsigned*)(Wf + (size_t)j * FDIM);
        float sum = 0.f;
        #pragma unroll 4
        for (int t = 0; t < FDIM / 64; ++t) {
            const unsigned za = zr[t * 32 + lane];
            const unsigned wa = wr[t * 32 + lane];
            const float zl = __uint_as_float(za << 16);
            const float zh = __uint_as_float(za & 0xffff0000u);
            const float wl = __uint_as_float(wa << 16);
            const float wh = __uint_as_float(wa & 0xffff0000u);
            sum = fmaf(zl, wl, sum);
            sum = fmaf(zh, wh, sum);
        }
        #pragma unroll
        for (int off = 16; off > 0; off >>= 1) sum += __shfl_xor(sum, off, 32);
        if (lane == 0)
            wred[wid] = avec[i] * bvec[j] * (s_p0 + sum) + gvec[i] + dvec[j];
    }
    __syncthreads();
    if (tid == 0) {
        float tot = 0.f;
        for (int k = 0; k < 8; ++k) tot += wred[k];
        part[blockIdx.x] = tot;  // +z_pdist2 part
    }
}

// Deterministic final reduction: fixed-order strided sums + LDS tree.
__global__ __launch_bounds__(256) void final_reduce_kernel(
    const float* __restrict__ part, int n, float* __restrict__ out)
{
    __shared__ float red[256];
    float s = 0.f;
    for (int i = threadIdx.x; i < n; i += 256) s += part[i];
    red[threadIdx.x] = s; __syncthreads();
    for (int st = 128; st > 0; st >>= 1) { if (threadIdx.x < st) red[threadIdx.x] += red[threadIdx.x + st]; __syncthreads(); }
    if (threadIdx.x == 0) out[0] = red[0];
}

extern "C" void kernel_launch(void* const* d_in, const int* in_sizes, int n_in,
                              void* d_out, int out_size, void* d_ws, size_t ws_size,
                              hipStream_t stream)
{
    (void)n_in; (void)out_size; (void)ws_size;
    const float* us        = (const float*)d_in[0];
    const float* vs        = (const float*)d_in[1];
    const float* gamma     = (const float*)d_in[2];
    const float* delta     = (const float*)d_in[3];
    const float* latent_z1 = (const float*)d_in[4];
    const float* latent_w1 = (const float*)d_in[5];
    const float* p_ks      = (const float*)d_in[6];
    const float* L_        = (const float*)d_in[7];
    const int*   sis       = (const int*)d_in[8];
    const int*   sjs       = (const int*)d_in[9];
    float* out = (float*)d_out;

    const int E = in_sizes[8] / LAYERS;
    const int NG = (N1 / BM) * (N2 / BN);       // gemm blocks per layer
    const int NS = (E + 7) / 8;                 // sparse blocks per layer

    char* w = (char*)d_ws;
    __hip_bfloat16* Zf = (__hip_bfloat16*)w;
    __hip_bfloat16* Wf = (__hip_bfloat16*)(w + (size_t)N1 * FDIM * 2);
    float* avec  = (float*)(w + (size_t)(N1 + N2) * FDIM * 2);
    float* bvec  = avec + N1;
    float* gvec  = bvec + N2;
    float* dvec  = gvec + N1;
    float* partG = dvec + N2;                   // LAYERS * NG
    float* partS = partG + LAYERS * NG;         // LAYERS * NS
    const int nPart = LAYERS * (NG + NS);

    for (int layer = 0; layer < LAYERS; ++layer) {
        feat_z_kernel<<<N1, 256, 0, stream>>>(us + (size_t)layer * N1 * DD,
                                              latent_z1, gamma, L_, Zf, avec, gvec, layer);
        feat_w_kernel<<<N2, 256, 0, stream>>>(vs + (size_t)layer * N2 * DD,
                                              latent_w1, delta, p_ks, Wf, bvec, dvec, layer);
        gemm_softplus_kernel<<<dim3(N2 / BN, N1 / BM), 256, 0, stream>>>(
            Zf, Wf, avec, bvec, gvec, dvec, p_ks, layer, partG + (size_t)layer * NG);
        sparse_kernel<<<NS, 256, 0, stream>>>(
            Zf, Wf, avec, bvec, gvec, dvec,
            sis + (size_t)layer * E, sjs + (size_t)layer * E,
            p_ks, layer, E, partS + (size_t)layer * NS);
    }
    final_reduce_kernel<<<1, 256, 0, stream>>>(partG, nPart, out);
}
